// RNN_25409026523649
// MI455X (gfx1250) — matrix operand order repaired and checked by reading, not compiled, not run on hardware
//
#include <hip/hip_runtime.h>
#include <hip/hip_bf16.h>
#include <math.h>

// ---------------------------------------------------------------------------
// Leaky-tanh RNN scan on gfx1250.
//   h' = (1-a)h + a(x W_in^T + tanh(h) W_hh^T + b_in + b_hh) + noise
//   out = clip(h' W_out^T + b_out, +-20)
//
// Key structure: row-panels of h are independent recurrences. 64 WGs =
// 8 groups x 8 WGs; group g owns batch rows [16g, 16g+16). Per step:
//   Phase A (WG-local): tanh(h rows) -> LDS bf16 panel, x_t rows -> LDS;
//                       wave 7 also does the output GEMM of step t-1.
//   Phase B: each wave v_wmma_f32_16x16x32_bf16 over its 16x16 tile,
//            A-fragments from LDS (ds_load_b128, bank-conflict-free via
//            1032-element padded rows), B-fragments (weights) from L2.
//   One 8-WG sibling barrier per step (agent-scope atomic + s_sleep).
// h and bf16(h_new) are double-buffered so sibling WGs may drift one step.
// ---------------------------------------------------------------------------

typedef __bf16 bf16;
typedef __attribute__((ext_vector_type(16))) __bf16 v16bf;
typedef __attribute__((ext_vector_type(8)))  __bf16 v8bf;
typedef __attribute__((ext_vector_type(8)))  float  v8f;

#define B_      128
#define H_      1024
#define NIN_    128
#define NOUT_   128
#define T_      512
#define NWG     64
#define NTHR    256
#define GRP_WGS 8
#define ALPHA_  0.25f
#define NSCALE_ 0.025f       // SIGMA_NEU * sqrt(ALPHA)

#define LDA_P   1032         // padded LDS row stride (bf16): 516 DW = 4 mod 64 banks
#define LDX_P   136          // padded LDS row stride for x panel: 68 DW = 4 mod 64

// workspace layout (byte offsets)
#define WS_BAR  0                          // 8 counters, 64B apart
#define WS_WIN  512                        // 1024*128  bf16
#define WS_WHH  (WS_WIN  + 131072 * 2)     // 1024*1024 bf16
#define WS_WOUT (WS_WHH  + 1048576 * 2)    // 128*1024  bf16
#define WS_HN0  (WS_WOUT + 131072 * 2)     // h_new bf16 buf0 [128x1024]
#define WS_HN1  (WS_HN0  + 131072 * 2)     // h_new bf16 buf1
#define WS_H0   (WS_HN1  + 131072 * 2)     // h fp32 buf0
#define WS_H1   (WS_H0   + 131072 * 4)     // h fp32 buf1

// ---- 8-WG sibling barrier (monotonic counter, agent scope) -----------------
__device__ __forceinline__ void sbar(unsigned* cnt, unsigned target) {
  __builtin_amdgcn_fence(__ATOMIC_RELEASE, "agent");
  __syncthreads();
  if (threadIdx.x == 0) {
    __hip_atomic_fetch_add(cnt, 1u, __ATOMIC_RELAXED, __HIP_MEMORY_SCOPE_AGENT);
    while (__hip_atomic_load(cnt, __ATOMIC_RELAXED, __HIP_MEMORY_SCOPE_AGENT) < target)
      __builtin_amdgcn_s_sleep(2);
  }
  __syncthreads();
  __builtin_amdgcn_fence(__ATOMIC_ACQUIRE, "agent");
}

// ---- fast tanh: 1 v_exp + 1 v_rcp, clamped so exp never overflows ----------
__device__ __forceinline__ float fast_tanh(float x) {
  x = fminf(10.0f, fmaxf(-10.0f, x));
  const float e = __expf(2.0f * x);
  return (e - 1.0f) * __frcp_rn(e + 1.0f);
}

// ---- one 16x16 tile, acc += A[16xK] * B^T (B row-major [N x K]) ------------
// ISA fragment layouts (wave32):
//   A: lane L -> row (L&15), K-chunks (L>>4)*8 and +16  (2x 16B loads)
//   B: lane L -> col (L&15) == weight row, K-chunks (L>>4)*8 and +16 (2x 16B loads)
// A may live in LDS (addrspace inferred -> ds_load_b128) or global.
__device__ __forceinline__ v8f wmma_acc(const bf16* __restrict__ A, int lda,
                                        const bf16* __restrict__ Bm, int ldb,
                                        int K, v8f acc, int lane) {
  const bf16* ap = A  + (lane & 15) * lda + ((lane >> 4) << 3);
  const bf16* bp = Bm + (lane & 15) * ldb + ((lane >> 4) << 3);
#pragma unroll 4
  for (int k = 0; k < K; k += 32) {
    v8bf  alo = *(const v8bf*)(ap + k);
    v8bf  ahi = *(const v8bf*)(ap + k + 16);
    v16bf av  = __builtin_shufflevector(alo, ahi, 0,1,2,3,4,5,6,7,8,9,10,11,12,13,14,15);
    v16bf bv  = __builtin_shufflevector(*(const v8bf*)(bp + k), *(const v8bf*)(bp + k + 16), 0,1,2,3,4,5,6,7,8,9,10,11,12,13,14,15);
    acc = __builtin_amdgcn_wmma_f32_16x16x32_bf16(false, av, false, bv,
                                                  (short)0, acc, false, false);
  }
  return acc;
}

// ---- output GEMM tile for step t: clip(h_new W_out^T + b) ------------------
__device__ __forceinline__ void out_gemm(int t, int m0, int n0, int lane,
                                         const bf16* __restrict__ hn_bf,
                                         const bf16* __restrict__ w_out_bf,
                                         const float* __restrict__ w_out_b,
                                         float* __restrict__ out_out) {
  v8f acc = {};
  acc = wmma_acc(hn_bf + m0 * H_, H_, w_out_bf + n0 * H_, H_, H_, acc, lane);
  const int rr = m0 + 8 * (lane >> 4);
  const int nn = n0 + (lane & 15);
  const float bias = w_out_b[nn];
#pragma unroll
  for (int r = 0; r < 8; ++r) {
    float v = acc[r] + bias;
    v = fminf(20.0f, fmaxf(-20.0f, v));
    out_out[((size_t)(rr + r) * T_ + t) * NOUT_ + nn] = v;
  }
}

// ---- prep: fp32 -> bf16 weights, init h buf0, reset group barriers ---------
__global__ __launch_bounds__(256) void rnn_prep(const float* __restrict__ w_in,
                                                const float* __restrict__ w_hh,
                                                const float* __restrict__ w_out,
                                                const float* __restrict__ h0,
                                                char* __restrict__ ws) {
  bf16* w_in_bf  = (bf16*)(ws + WS_WIN);
  bf16* w_hh_bf  = (bf16*)(ws + WS_WHH);
  bf16* w_out_bf = (bf16*)(ws + WS_WOUT);
  float* hbuf0   = (float*)(ws + WS_H0);
  unsigned* bars = (unsigned*)(ws + WS_BAR);
  const int g = blockIdx.x * 256 + threadIdx.x;
  const int s = gridDim.x * 256;
  if (g < GRP_WGS) bars[g * 16] = 0u;
  for (int i = g; i < H_ * NIN_;  i += s) w_in_bf[i]  = (bf16)w_in[i];
  for (int i = g; i < H_ * H_;    i += s) w_hh_bf[i]  = (bf16)w_hh[i];
  for (int i = g; i < NOUT_ * H_; i += s) w_out_bf[i] = (bf16)w_out[i];
  for (int i = g; i < B_ * H_;    i += s) hbuf0[i]    = h0[i];
}

// ---- persistent scan kernel ------------------------------------------------
__global__ __launch_bounds__(NTHR) void rnn_main(
    const float* __restrict__ xsig,   // [B, T, NIN]
    const float* __restrict__ noise,  // [T, B, H] raw randn
    const float* __restrict__ w_in_b, const float* __restrict__ w_hh_b,
    const float* __restrict__ w_out_b,
    char* __restrict__ ws,
    float* __restrict__ hid_out,      // [B, T, H]
    float* __restrict__ out_out,      // [B, T, NOUT]
    float* __restrict__ hfin) {       // [B, H]
  __shared__ bf16 lds_a[16 * LDA_P];  // tanh(h) panel, padded rows
  __shared__ bf16 lds_x[16 * LDX_P];  // x_t panel, padded rows

  const bf16* w_in_bf  = (const bf16*)(ws + WS_WIN);
  const bf16* w_hh_bf  = (const bf16*)(ws + WS_WHH);
  const bf16* w_out_bf = (const bf16*)(ws + WS_WOUT);
  bf16*  hn0   = (bf16*)(ws + WS_HN0);
  bf16*  hn1   = (bf16*)(ws + WS_HN1);
  float* hb0   = (float*)(ws + WS_H0);
  float* hb1   = (float*)(ws + WS_H1);

  const int tid    = threadIdx.x;
  const int lane   = tid & 31;
  const int wave   = tid >> 5;
  const int group  = blockIdx.x >> 3;          // 0..7: batch-row panel
  const int member = blockIdx.x & 7;           // 0..7 within group
  const int m0     = group * 16;
  const int nt     = member * 8 + wave;        // 0..63: hidden tile
  const int n0     = nt * 16;
  unsigned* bar    = (unsigned*)(ws + WS_BAR) + group * 16;
  unsigned epoch = 0;

  for (int t = 0; t < T_; ++t) {
    const float* hsrc = (t & 1) ? hb1 : hb0;
    float*       hdst = (t & 1) ? hb0 : hb1;
    bf16*        hn_w = (t & 1) ? hn1 : hn0;
    const bf16*  hn_r = (t & 1) ? hn0 : hn1;   // written at step t-1

    // ---- Phase A (WG-local): tanh(h rows) -> LDS, x_t rows -> LDS ---------
    for (int v = tid; v < 16 * 256; v += NTHR) {          // 16 rows x 256 col4
      const int r = v >> 8;
      const int c = (v & 255) << 2;
      const float4 hv = *(const float4*)(hsrc + (size_t)(m0 + r) * H_ + c);
      bf16* dst = &lds_a[r * LDA_P + c];
      dst[0] = (bf16)fast_tanh(hv.x); dst[1] = (bf16)fast_tanh(hv.y);
      dst[2] = (bf16)fast_tanh(hv.z); dst[3] = (bf16)fast_tanh(hv.w);
    }
    for (int v = tid; v < 16 * 32; v += NTHR) {           // 16 rows x 32 col4
      const int r = v >> 5;
      const int c = (v & 31) << 2;
      const float4 xv = *(const float4*)(xsig + ((size_t)(m0 + r) * T_ + t) * NIN_ + c);
      bf16* dst = &lds_x[r * LDX_P + c];
      dst[0] = (bf16)xv.x; dst[1] = (bf16)xv.y;
      dst[2] = (bf16)xv.z; dst[3] = (bf16)xv.w;
    }
    if (t > 0 && wave == 7)   // output GEMM of previous step (wave-uniform)
      out_gemm(t - 1, m0, member * 16, lane, hn_r, w_out_bf, w_out_b, out_out);
    __syncthreads();

    // ---- Phase B: step GEMM + leaky update --------------------------------
    v8f acc = {};
    acc = wmma_acc(lds_x, LDX_P, w_in_bf + n0 * NIN_, NIN_, NIN_, acc, lane);
    acc = wmma_acc(lds_a, LDA_P, w_hh_bf + n0 * H_,   H_,   H_,   acc, lane);

    const int rr = m0 + 8 * (lane >> 4);
    const int jj = n0 + (lane & 15);
    const float bias = w_in_b[jj] + w_hh_b[jj];
    const float* nz = noise + (size_t)t * B_ * H_;
#pragma unroll
    for (int r = 0; r < 8; ++r) {
      const int bb = rr + r;
      const size_t hidx = (size_t)bb * H_ + jj;
      const float tmp = acc[r] + bias;
      const float hn = (1.0f - ALPHA_) * hsrc[hidx] + ALPHA_ * tmp + NSCALE_ * nz[hidx];
      hdst[hidx] = hn;
      hid_out[((size_t)bb * T_ + t) * H_ + jj] = hn;
      hn_w[hidx] = (bf16)hn;
    }
    sbar(bar, ++epoch * GRP_WGS);
  }

  // ---- tail: output GEMM of last step + h_final ---------------------------
  if (wave == 7)
    out_gemm(T_ - 1, m0, member * 16, lane, hn1, w_out_bf, w_out_b, out_out);
  {
    const int idx = member * NTHR + tid;                  // 2048 threads/group
    for (int i = idx; i < 16 * H_; i += GRP_WGS * NTHR) {
      const int r = i >> 10, c = i & 1023;
      hfin[(size_t)(m0 + r) * H_ + c] = hb0[(size_t)(m0 + r) * H_ + c];
    }
  }
}

extern "C" void kernel_launch(void* const* d_in, const int* in_sizes, int n_in,
                              void* d_out, int out_size, void* d_ws, size_t ws_size,
                              hipStream_t stream) {
  (void)in_sizes; (void)n_in; (void)out_size; (void)ws_size;
  const float* xsig    = (const float*)d_in[0];
  const float* h0      = (const float*)d_in[1];
  const float* noise   = (const float*)d_in[2];
  const float* w_in    = (const float*)d_in[3];
  const float* w_in_b  = (const float*)d_in[4];
  const float* w_hh    = (const float*)d_in[5];
  const float* w_hh_b  = (const float*)d_in[6];
  const float* w_out   = (const float*)d_in[7];
  const float* w_out_b = (const float*)d_in[8];
  char* ws = (char*)d_ws;

  float* hid_out = (float*)d_out;
  float* out_out = hid_out + (size_t)B_ * T_ * H_;
  float* hfin    = out_out + (size_t)B_ * T_ * NOUT_;

  rnn_prep<<<512, 256, 0, stream>>>(w_in, w_hh, w_out, h0, ws);
  rnn_main<<<NWG, NTHR, 0, stream>>>(xsig, noise, w_in_b, w_hh_b, w_out_b,
                                     ws, hid_out, out_out, hfin);
}
